// MultiHeadAttention_19189913878579
// MI455X (gfx1250) — matrix operand order repaired and checked by reading, not compiled, not run on hardware
//
#include <hip/hip_runtime.h>
#include <math.h>

// ---------------------------------------------------------------------------
// MHA forward for MI455X (gfx1250, wave32, WMMA + TDM).
// All matmuls in bf16 via v_wmma_f32_16x16x32_bf16 with fp32 accumulation.
// Flash-attention fusion; K/V tiles staged into LDS by the Tensor Data Mover
// (tensor_load_to_lds, TENSORcnt), double-buffered.
// ---------------------------------------------------------------------------

typedef __bf16 bf16;
typedef bf16  v16bf __attribute__((ext_vector_type(16)));
typedef bf16  v8bf  __attribute__((ext_vector_type(8)));
typedef float v8f   __attribute__((ext_vector_type(8)));

typedef unsigned int u32x4 __attribute__((ext_vector_type(4)));
typedef int          i32x4 __attribute__((ext_vector_type(4)));
typedef int          i32x8 __attribute__((ext_vector_type(8)));

#define NHEAD  8
#define DMODEL 512
#define DKV    64
#define BATCH  4
#define SEQ    2048

// ---- WMMA wrapper ---------------------------------------------------------
__device__ __forceinline__ v8f wmma_bf16(v16bf a, v16bf b, v8f c) {
  // (neg_a, A, neg_b, B, c_mod, C, reuse_a, reuse_b)
  return __builtin_amdgcn_wmma_f32_16x16x32_bf16(false, a, false, b,
                                                 (short)0, c, false, false);
}

// ---- A operand: 16x32 bf16, row-major source, stride ld (elements) --------
// ISA layout: lanes 0-15 hold row M=lane, K=0..7 (v0-3) and K=16..23 (v4-7);
// lanes 16-31 hold row M=lane-16, K=8..15 and K=24..31.
__device__ __forceinline__ v16bf load_a_tile(const bf16* base, int ld) {
  int lane = threadIdx.x & 31;
  int r    = lane & 15;
  int half = lane >> 4;
  const bf16* p = base + (long)r * ld + half * 8;
  v8bf lo = *(const v8bf*)(p);        // K = half*8 .. half*8+7
  v8bf hi = *(const v8bf*)(p + 16);   // K = 16+half*8 ..
  return __builtin_shufflevector(lo, hi, 0,1,2,3,4,5,6,7,8,9,10,11,12,13,14,15);
}

// ---- B operand: 32x16 bf16 where B(k,n) = Bt[n][k], Bt row-major ----------
// ISA layout: lanes 0-15 hold col N=lane, K=0..7 and K=16..23; lanes 16-31
// col N=lane-16, K=8..15 and K=24..31 (same k order as the A operand).
__device__ __forceinline__ v16bf load_bt_tile(const bf16* base, int ld) {
  int lane = threadIdx.x & 31;
  int n    = lane & 15;
  int half = lane >> 4;
  const bf16* p = base + (long)n * ld + half * 8;
  v8bf lo = *(const v8bf*)(p);
  v8bf hi = *(const v8bf*)(p + 16);
  return __builtin_shufflevector(lo, hi, 0,1,2,3,4,5,6,7,8,9,10,11,12,13,14,15);
}

// ---- wave-half reductions (C layout keeps one output row in lanes L%16) ---
__device__ __forceinline__ float red_max16(float v) {
#pragma unroll
  for (int m = 1; m <= 8; m <<= 1) v = fmaxf(v, __shfl_xor(v, m, 32));
  return v;
}
__device__ __forceinline__ float red_sum16(float v) {
#pragma unroll
  for (int m = 1; m <= 8; m <<= 1) v += __shfl_xor(v, m, 32);
  return v;
}

// ---- TDM: 2D tile (tile_d0 x tile_d1, bf16) from global -> LDS ------------
// Descriptor per CDNA5 ISA ch.8 (D#).  Tensor dims are set equal to the tile
// so nothing is out of bounds; stride0 (elements) walks the source rows.
// 6-arg builtin form: (g0 u32x4, g1 i32x8, g2 i32x4, g3 i32x4, i32x8, cpol).
__device__ __forceinline__ void tdm_load_2d(unsigned lds_off, const void* gaddr,
                                            unsigned tile_d0, unsigned tile_d1,
                                            unsigned stride0) {
  unsigned long long ga = (unsigned long long)gaddr;
  u32x4 g0;
  g0[0] = 1u;                                          // count=1, user desc
  g0[1] = lds_off;                                     // lds_addr (bytes)
  g0[2] = (unsigned)(ga & 0xffffffffu);                // global_addr[31:0]
  g0[3] = (unsigned)((ga >> 32) & 0x1ffffffu)          // global_addr[56:32]
        | (2u << 30);                                  // type = 2 ("image")
  i32x8 g1;
  g1[0] = (int)(1u << 16);                             // data_size=1 (2 bytes)
  g1[1] = (int)((tile_d0 & 0xffffu) << 16);            // tensor_dim0[15:0]
  g1[2] = (int)((tile_d1 & 0xffffu) << 16);            // d0 hi=0 | tensor_dim1 lo
  g1[3] = (int)((tile_d0 & 0xffffu) << 16);            // d1 hi=0 | tile_dim0
  g1[4] = (int)(tile_d1 & 0xffffu);                    // tile_dim1 | tile_dim2=0
  g1[5] = (int)stride0;                                // tensor_dim0_stride lo
  g1[6] = 0;                                           // stride0 hi | dim1_stride lo
  g1[7] = 0;
  i32x4 z4 = {0, 0, 0, 0};                             // groups 2/3 unused (2D)
  i32x8 z8 = {0, 0, 0, 0, 0, 0, 0, 0};
  __builtin_amdgcn_tensor_load_to_lds(g0, g1, z4, z4, z8, 0);
}

// ---------------------------------------------------------------------------
// Prep kernels
// ---------------------------------------------------------------------------
__global__ __launch_bounds__(256) void cvt_f32_bf16(const float* __restrict__ in,
                                                    bf16* __restrict__ out, long n) {
  long i = (long)blockIdx.x * 256 + threadIdx.x;
  if (i < n) out[i] = (bf16)in[i];
}

// out[b][c][r] = (bf16) in[b][r][c]
__global__ __launch_bounds__(256) void transpose_cvt(const float* __restrict__ in,
                                                     bf16* __restrict__ out,
                                                     int rows, int cols) {
  long idx = (long)blockIdx.x * 256 + threadIdx.x;
  long n = (long)rows * cols;
  if (idx >= n) return;
  int r = (int)(idx / cols), c = (int)(idx % cols);
  const float* ib = in + (long)blockIdx.y * n;
  bf16* ob = out + (long)blockIdx.y * n;
  ob[(long)c * rows + r] = (bf16)ib[idx];
}

// ---------------------------------------------------------------------------
// Generic WMMA GEMM: C[row, n] = sum_k A[row,k] * Bt[n,k]
// block = 256 thr (8 waves); block tile 128 rows x 64 cols; wave = 16 x 64.
// Output address (bytes): (row>>11)*cBS + (row&2047)*cRS + z*cZS + n*cCS
// (S=2048 hardcoded in the row split).  storeF32 selects f32 vs bf16 store.
// ---------------------------------------------------------------------------
__global__ __launch_bounds__(256)
void wmma_gemm_kernel(const bf16* __restrict__ A, int lda,
                      const bf16* __restrict__ Bt, int ldb, long bZS,
                      char* __restrict__ C,
                      long cBS, long cRS, long cZS, long cCS,
                      int K, int storeF32)
{
  int wave = threadIdx.x >> 5;
  int lane = threadIdx.x & 31;
  long row0 = (long)blockIdx.x * 128 + wave * 16;
  int  n0   = blockIdx.y * 64;
  int  z    = blockIdx.z;

  const bf16* Ab = A + row0 * lda;
  const bf16* Bb = Bt + (long)z * bZS + (long)n0 * ldb;

  v8f acc[4] = {};
  for (int k0 = 0; k0 < K; k0 += 32) {
    v16bf a = load_a_tile(Ab + k0, lda);
#pragma unroll
    for (int t = 0; t < 4; ++t) {
      v16bf b = load_bt_tile(Bb + (long)(t * 16) * ldb + k0, ldb);
      acc[t] = wmma_bf16(a, b, acc[t]);
    }
  }

  int col = lane & 15, half = lane >> 4;
#pragma unroll
  for (int t = 0; t < 4; ++t) {
#pragma unroll
    for (int r = 0; r < 8; ++r) {
      long row = row0 + r + 8 * half;
      long n   = n0 + t * 16 + col;
      char* p = C + (row >> 11) * cBS + (row & 2047) * cRS + (long)z * cZS + n * cCS;
      if (storeF32) *(float*)p = acc[t][r];
      else          *(bf16*)p  = (bf16)acc[t][r];
    }
  }
}

// ---------------------------------------------------------------------------
// Flash attention: grid (S/128, H, B), 256 thr (8 waves), wave = 16 q-rows.
// qh, kh: [B,H,S,64] bf16 row-major;  vTh: [B,H,64,S] bf16 row-major.
// K/V tiles staged into LDS by the TDM (wave 0 issues, double-buffered).
// o: [B,S,H*64] bf16 (heads concatenated, ready for the WO GEMM).
// ---------------------------------------------------------------------------
__global__ __launch_bounds__(256)
void flash_attn_kernel(const bf16* __restrict__ qh, const bf16* __restrict__ kh,
                       const bf16* __restrict__ vTh, bf16* __restrict__ o)
{
  __shared__ bf16 ldsK[2][32 * DKV];  // 32 keys x 64 dk, double-buffered (8KB)
  __shared__ bf16 ldsV[2][DKV * 32];  // 64 vdim x 32 keys, double-buffered (8KB)
  __shared__ bf16 pLds[8][16 * 32];   // wave-private 16x32 P staging (8KB)

  int wave = threadIdx.x >> 5, lane = threadIdx.x & 31;
  int h = blockIdx.y, b = blockIdx.z;
  int q0 = blockIdx.x * 128 + wave * 16;
  long bh = (long)b * NHEAD + h;

  const bf16* Q  = qh  + bh * (long)SEQ * DKV;
  const bf16* Km = kh  + bh * (long)SEQ * DKV;
  const bf16* Vt = vTh + bh * (long)DKV * SEQ;

  v16bf aq0 = load_a_tile(Q + (long)q0 * DKV,      DKV);
  v16bf aq1 = load_a_tile(Q + (long)q0 * DKV + 32, DKV);

  float m[8], l[8];
#pragma unroll
  for (int r = 0; r < 8; ++r) { m[r] = -1e30f; l[r] = 0.f; }
  v8f oacc[4] = {};

  const float sc = 0.125f * 1.44269504088896f;   // 1/sqrt(dk) * log2(e)

  // prologue: TDM-stage tile 0 (wave 0 only; EXEC-independent, SGPR descriptor)
  if (wave == 0) {
    tdm_load_2d((unsigned)(unsigned long long)&ldsK[0][0], Km, DKV, 32, DKV);
    tdm_load_2d((unsigned)(unsigned long long)&ldsV[0][0], Vt, 32, DKV, SEQ);
  }

  const int nIter = SEQ / 32;
  for (int it = 0; it < nIter; ++it) {
    int s0  = it * 32;
    int cur = it & 1, nxt = cur ^ 1;

    if (wave == 0) __builtin_amdgcn_s_wait_tensorcnt(0);
    __syncthreads();                          // tile `it` visible to all waves

    if (wave == 0 && it + 1 < nIter) {        // prefetch tile it+1 via TDM
      int s1 = s0 + 32;
      tdm_load_2d((unsigned)(unsigned long long)&ldsK[nxt][0],
                  Km + (long)s1 * DKV, DKV, 32, DKV);
      tdm_load_2d((unsigned)(unsigned long long)&ldsV[nxt][0],
                  Vt + s1, 32, DKV, SEQ);
    }

    const bf16* Kt = &ldsK[cur][0];           // [32 keys][64] row-major
    const bf16* Vv = &ldsV[cur][0];           // [64 vdim][32 keys] row-major

    // ---- scores: 16 q x 32 keys, dk=64 folded as 2 x K32 -----------------
    v8f sA = {}, sB = {};
    {
      v16bf b00 = load_bt_tile(Kt,                DKV);
      v16bf b01 = load_bt_tile(Kt + 32,           DKV);
      sA = wmma_bf16(aq0, b00, sA);
      sA = wmma_bf16(aq1, b01, sA);
      v16bf b10 = load_bt_tile(Kt + 16 * DKV,      DKV);
      v16bf b11 = load_bt_tile(Kt + 16 * DKV + 32, DKV);
      sB = wmma_bf16(aq0, b10, sB);
      sB = wmma_bf16(aq1, b11, sB);
    }

    // ---- online softmax (exp2 domain) ------------------------------------
    float mnew[8], alpha[8];
#pragma unroll
    for (int r = 0; r < 8; ++r) {
      float t = fmaxf(sA[r], sB[r]);
      t = red_max16(t) * sc;                 // sc > 0: max commutes with scale
      mnew[r]  = fmaxf(m[r], t);
      alpha[r] = exp2f(m[r] - mnew[r]);
    }
    v8f p0, p1;
#pragma unroll
    for (int r = 0; r < 8; ++r) {
      p0[r] = exp2f(sA[r] * sc - mnew[r]);
      p1[r] = exp2f(sB[r] * sc - mnew[r]);
      l[r]  = l[r] * alpha[r] + red_sum16(p0[r] + p1[r]);
      m[r]  = mnew[r];
    }
#pragma unroll
    for (int t4 = 0; t4 < 4; ++t4)
#pragma unroll
      for (int r = 0; r < 8; ++r) oacc[t4][r] *= alpha[r];

    // ---- P: C layout -> A layout via wave-private LDS ---------------------
    bf16* pw = &pLds[wave][0];
    int colb = lane & 15, halfb = lane >> 4;
    __builtin_amdgcn_wave_barrier();
#pragma unroll
    for (int r = 0; r < 8; ++r) {
      int row = r + 8 * halfb;
      pw[row * 32 + colb]      = (bf16)p0[r];
      pw[row * 32 + colb + 16] = (bf16)p1[r];
    }
    __builtin_amdgcn_wave_barrier();           // LDS is in-order within a wave
    v16bf ap = load_a_tile(pw, 32);

    // ---- O += P (16x32) x V (32 keys x 64), B from transposed V -----------
#pragma unroll
    for (int t4 = 0; t4 < 4; ++t4) {
      v16bf bv = load_bt_tile(Vv + (long)(t4 * 16) * 32, 32);
      oacc[t4] = wmma_bf16(ap, bv, oacc[t4]);
    }

    __syncthreads();   // all waves done with buffer `cur` before it is reloaded
  }

  // ---- epilogue: normalize and store bf16 heads-concatenated --------------
  int col = lane & 15, half = lane >> 4;
#pragma unroll
  for (int r = 0; r < 8; ++r) {
    float inv = 1.0f / l[r];
    long row  = q0 + r + 8 * half;
    long base = ((long)b * SEQ + row) * (NHEAD * DKV) + h * DKV;
#pragma unroll
    for (int t4 = 0; t4 < 4; ++t4)
      o[base + t4 * 16 + col] = (bf16)(oacc[t4][r] * inv);
  }
}

// ---------------------------------------------------------------------------
// Host launcher
// ---------------------------------------------------------------------------
extern "C" void kernel_launch(void* const* d_in, const int* in_sizes, int n_in,
                              void* d_out, int out_size, void* d_ws, size_t ws_size,
                              hipStream_t stream)
{
  const float* Q  = (const float*)d_in[0];
  const float* K  = (const float*)d_in[1];
  const float* V  = (const float*)d_in[2];
  const float* WQ = (const float*)d_in[3];
  const float* WK = (const float*)d_in[4];
  const float* WV = (const float*)d_in[5];
  const float* WO = (const float*)d_in[6];

  char* ws = (char*)d_ws;
  size_t off = 0;
  auto alloc = [&](size_t bytes) -> char* {
    char* p = ws + off;
    off = (off + bytes + 255) & ~(size_t)255;
    return p;
  };

  const size_t nQKV = (size_t)BATCH * SEQ * DMODEL;          // 4M elements
  bf16* Qb  = (bf16*)alloc(nQKV * 2);
  bf16* Kb  = (bf16*)alloc(nQKV * 2);
  bf16* Vb  = (bf16*)alloc(nQKV * 2);
  bf16* WQt = (bf16*)alloc((size_t)NHEAD * DKV * DMODEL * 2);
  bf16* WKt = (bf16*)alloc((size_t)NHEAD * DKV * DMODEL * 2);
  bf16* WVt = (bf16*)alloc((size_t)NHEAD * DKV * DMODEL * 2);
  bf16* WOt = (bf16*)alloc((size_t)DMODEL * DMODEL * 2);
  bf16* qh  = (bf16*)alloc((size_t)BATCH * NHEAD * SEQ * DKV * 2);
  bf16* kh  = (bf16*)alloc((size_t)BATCH * NHEAD * SEQ * DKV * 2);
  bf16* vT  = (bf16*)alloc((size_t)BATCH * NHEAD * DKV * SEQ * 2);
  bf16* ob  = Qb;   // Qb is dead after the q projection: reuse for attn output

  dim3 blk(256);

  // 1) convert activations to bf16
  int nCvt = (int)((nQKV + 255) / 256);
  cvt_f32_bf16<<<nCvt, blk, 0, stream>>>(Q, Qb, (long)nQKV);
  cvt_f32_bf16<<<nCvt, blk, 0, stream>>>(K, Kb, (long)nQKV);
  cvt_f32_bf16<<<nCvt, blk, 0, stream>>>(V, Vb, (long)nQKV);

  // 2) transpose+convert weights (Bt layout for contiguous WMMA B loads)
  transpose_cvt<<<dim3(128, NHEAD), blk, 0, stream>>>(WQ, WQt, DMODEL, DKV);
  transpose_cvt<<<dim3(128, NHEAD), blk, 0, stream>>>(WK, WKt, DMODEL, DKV);
  transpose_cvt<<<dim3(128, NHEAD), blk, 0, stream>>>(WV, WVt, DMODEL, DKV);
  transpose_cvt<<<dim3(1024, 1),    blk, 0, stream>>>(WO, WOt, DMODEL, DMODEL);

  // 3) projections  (grid.x: 8192 rows / 128; grid.z: head)
  {
    long cBS = (long)NHEAD * SEQ * DKV * 2;   // per-batch jump in [B,H,S,64]
    long cRS = (long)DKV * 2;
    long cZS = (long)SEQ * DKV * 2;
    wmma_gemm_kernel<<<dim3(64, 1, NHEAD), blk, 0, stream>>>(
        Qb, DMODEL, WQt, DMODEL, (long)DKV * DMODEL,
        (char*)qh, cBS, cRS, cZS, 2, DMODEL, 0);
    wmma_gemm_kernel<<<dim3(64, 1, NHEAD), blk, 0, stream>>>(
        Kb, DMODEL, WKt, DMODEL, (long)DKV * DMODEL,
        (char*)kh, cBS, cRS, cZS, 2, DMODEL, 0);
    // v stored transposed: [B,H,64,S]  (row index = s -> stride 2 bytes,
    // col index = v-dim -> stride S*2 bytes)
    wmma_gemm_kernel<<<dim3(64, 1, NHEAD), blk, 0, stream>>>(
        Vb, DMODEL, WVt, DMODEL, (long)DKV * DMODEL,
        (char*)vT, (long)NHEAD * DKV * SEQ * 2, 2, (long)DKV * SEQ * 2,
        (long)SEQ * 2, DMODEL, 0);
  }

  // 4) fused flash attention (TDM-staged K/V)
  flash_attn_kernel<<<dim3(SEQ / 128, NHEAD, BATCH), blk, 0, stream>>>(qh, kh, vT, ob);

  // 5) output projection -> d_out (fp32)
  wmma_gemm_kernel<<<dim3(64, 8, 1), blk, 0, stream>>>(
      ob, DMODEL, WOt, DMODEL, 0,
      (char*)d_out, (long)SEQ * DMODEL * 4, (long)DMODEL * 4, 0, 4, DMODEL, 1);
}
